// ReasoningBlock_8177617732086
// MI455X (gfx1250) — matrix operand order repaired and checked by reading, not compiled, not run on hardware
//
#include <hip/hip_runtime.h>
#include <hip/hip_bf16.h>
#include <math.h>

// ---------------------------------------------------------------------------
// Transformer block (MHA + SwiGLU, RMSNorm) for MI455X / gfx1250.
// bf16 WMMA GEMMs (v_wmma_f32_16x16x32_bf16), fp32 softmax/norms.
// Double-buffered LDS staging via async global->LDS copies when available.
// Shapes: TOK=65536 tokens (B=16384 x S=4), D=1024, DH=4096, H=16, dh=64.
// ---------------------------------------------------------------------------

typedef __attribute__((ext_vector_type(16))) __bf16 v16bf;
typedef __attribute__((ext_vector_type(8)))  float  v8f;
typedef __attribute__((ext_vector_type(4)))  int    v4i;

union FragBf { v16bf v; uint4 q[2]; };

#if __has_builtin(__builtin_amdgcn_global_load_async_to_lds_b128) && \
    __has_builtin(__builtin_amdgcn_s_wait_asynccnt)
#define HAVE_ASYNC_LDS 1
#else
#define HAVE_ASYNC_LDS 0
#endif

#if HAVE_ASYNC_LDS
typedef __attribute__((address_space(1))) v4i gv4i;  // global int4
typedef __attribute__((address_space(3))) v4i sv4i;  // LDS int4
#endif

__device__ __forceinline__ void copy16_g2l(unsigned short* lds_dst,
                                           const unsigned short* gsrc) {
#if HAVE_ASYNC_LDS
  __builtin_amdgcn_global_load_async_to_lds_b128(
      (gv4i*)gsrc, (sv4i*)lds_dst, 0, 0);
#else
  *(uint4*)lds_dst = *(const uint4*)gsrc;
#endif
}

__device__ __forceinline__ void async_wait_all() {
#if HAVE_ASYNC_LDS
  __builtin_amdgcn_s_wait_asynccnt(0);
#endif
}

__device__ __forceinline__ unsigned short f2bf(float f) {
  unsigned u = __float_as_uint(f);
  u += 0x7FFFu + ((u >> 16) & 1u);          // round to nearest even
  return (unsigned short)(u >> 16);
}
__device__ __forceinline__ float bf2f(unsigned short h) {
  return __uint_as_float(((unsigned)h) << 16);
}

// ---------------------------------------------------------------------------
// fp32 -> bf16 conversion (weights + activations), grid-stride
// ---------------------------------------------------------------------------
__global__ void cvt_f32_to_bf16(const float* __restrict__ src,
                                unsigned short* __restrict__ dst, int n) {
  for (int i = blockIdx.x * blockDim.x + threadIdx.x; i < n;
       i += gridDim.x * blockDim.x)
    dst[i] = f2bf(src[i]);
}

// ---------------------------------------------------------------------------
// Tiled bf16 GEMM:  Dtile[M,N] = A[M,K] * B[N,K]^T (+bias)
// Block tile 64x128, 8 waves (2 x 4), wave tile 32x32 (2x2 WMMA frags).
// Double-buffered LDS; async copies overlap next tile with current WMMAs.
// EPI: 0 = store bf16, 1 = store fp32, 2 = dual-B silu(g)*u -> bf16
// ---------------------------------------------------------------------------
template <int EPI>
__global__ __launch_bounds__(256) void gemm_bf16_wmma(
    const unsigned short* __restrict__ A, int lda,
    const unsigned short* __restrict__ B, const unsigned short* __restrict__ B2,
    int ldb, void* __restrict__ Dp, int ldd, const float* __restrict__ bias,
    int N, int K) {
  constexpr int BM = 64, BN = 128, KT = 32;
  constexpr int LS = KT + 8;  // LDS row stride (bf16 elems), 80B: 16B aligned

  __shared__ unsigned short As[2][BM * LS];
  __shared__ unsigned short Bs[2][BN * LS];
  __shared__ unsigned short Bs2[2][(EPI == 2) ? BN * LS : 8];

  const int tid  = threadIdx.x;
  const int lane = tid & 31;
  const int wave = tid >> 5;
  const int wm = wave >> 2;              // 0..1
  const int wn = wave & 3;               // 0..3
  const int m0 = blockIdx.x * BM;
  const int n0 = blockIdx.y * BN;

  v8f acc[2][2];
  v8f acc2[2][2];
  for (int i = 0; i < 2; ++i)
    for (int j = 0; j < 2; ++j)
      for (int e = 0; e < 8; ++e) { acc[i][j][e] = 0.f; acc2[i][j][e] = 0.f; }

  // global->LDS staging assignments
  const int a_row = tid >> 2;            // 0..63
  const int a_col = (tid & 3) * 8;       // 0,8,16,24
  const int b_row = tid >> 1;            // 0..127
  const int b_col = (tid & 1) * 16;      // 0,16

  const int fm = lane & 15;              // frag row / col within 16
  const int kh = lane >> 4;              // K-half select

  const unsigned short* agp = &A[(size_t)(m0 + a_row) * lda + a_col];
  const unsigned short* bgp = &B[(size_t)(n0 + b_row) * ldb + b_col];
  const unsigned short* b2gp =
      (EPI == 2) ? &B2[(size_t)(n0 + b_row) * ldb + b_col] : nullptr;

  auto stage = [&](int k0, int buf) {
    copy16_g2l(&As[buf][a_row * LS + a_col], agp + k0);
    copy16_g2l(&Bs[buf][b_row * LS + b_col], bgp + k0);
    copy16_g2l(&Bs[buf][b_row * LS + b_col + 8], bgp + k0 + 8);
    if (EPI == 2) {
      copy16_g2l(&Bs2[buf][b_row * LS + b_col], b2gp + k0);
      copy16_g2l(&Bs2[buf][b_row * LS + b_col + 8], b2gp + k0 + 8);
    }
  };

  const int ktiles = K / KT;
  stage(0, 0);
  async_wait_all();
  __syncthreads();

  int buf = 0;
  for (int kt = 0; kt < ktiles; ++kt) {
    // kick off next tile into the other buffer (async: no VGPR round trip)
    if (kt + 1 < ktiles) stage((kt + 1) * KT, buf ^ 1);

    // compute on current buffer: fragment loads per ISA VGPR layouts
    const unsigned short* as = &As[buf][0];
    const unsigned short* bs = &Bs[buf][0];
    const unsigned short* bs2 = &Bs2[buf][0];

    FragBf af[2], bfr[2], bfr2[2];
    for (int fi = 0; fi < 2; ++fi) {
      const int r = wm * 32 + fi * 16 + fm;
      af[fi].q[0] = *(const uint4*)&as[r * LS + kh * 8];       // K 0..7 / 8..15
      af[fi].q[1] = *(const uint4*)&as[r * LS + 16 + kh * 8];  // K 16..23 / 24..31
    }
    for (int fj = 0; fj < 2; ++fj) {
      const int c = wn * 32 + fj * 16 + fm;
      bfr[fj].q[0] = *(const uint4*)&bs[c * LS + kh * 8];      // K 0..7 / 8..15
      bfr[fj].q[1] = *(const uint4*)&bs[c * LS + 16 + kh * 8];
      if (EPI == 2) {
        bfr2[fj].q[0] = *(const uint4*)&bs2[c * LS + kh * 8];
        bfr2[fj].q[1] = *(const uint4*)&bs2[c * LS + 16 + kh * 8];
      }
    }
    for (int fi = 0; fi < 2; ++fi)
      for (int fj = 0; fj < 2; ++fj) {
        acc[fi][fj] = __builtin_amdgcn_wmma_f32_16x16x32_bf16(
            false, af[fi].v, false, bfr[fj].v, (short)0, acc[fi][fj], false,
            false);
        if (EPI == 2)
          acc2[fi][fj] = __builtin_amdgcn_wmma_f32_16x16x32_bf16(
              false, af[fi].v, false, bfr2[fj].v, (short)0, acc2[fi][fj], false,
              false);
      }

    async_wait_all();      // staging for buf^1 landed in LDS
    __syncthreads();       // everyone done reading buf + writing buf^1
    buf ^= 1;
  }

  // epilogue: C/D layout -> VGPR r holds row (r + 8*(lane>=16)), col = lane&15
  const int roff = (lane >> 4) * 8;
  const int ccol = lane & 15;
  for (int fi = 0; fi < 2; ++fi)
    for (int fj = 0; fj < 2; ++fj)
      for (int r = 0; r < 8; ++r) {
        const int grow = m0 + wm * 32 + fi * 16 + roff + r;
        const int gcol = n0 + wn * 32 + fj * 16 + ccol;
        float v = acc[fi][fj][r];
        if (bias) v += bias[gcol];
        const size_t idx = (size_t)grow * ldd + gcol;
        if (EPI == 0) {
          ((unsigned short*)Dp)[idx] = f2bf(v);
        } else if (EPI == 1) {
          ((float*)Dp)[idx] = v;
        } else {  // silu(g) * u
          const float g = v;
          const float u = acc2[fi][fj][r];
          const float h = (g / (1.f + __expf(-g))) * u;
          ((unsigned short*)Dp)[idx] = f2bf(h);
        }
      }
}

// ---------------------------------------------------------------------------
// Attention: one block per batch (S=4 tokens, H=16 heads, dh=64).
// qkv rows are [q(1024) | k(1024) | v(1024)] bf16.
// ---------------------------------------------------------------------------
__global__ __launch_bounds__(256) void attention_kernel(
    const unsigned short* __restrict__ QKV, unsigned short* __restrict__ CTX) {
  __shared__ unsigned short qkvs[4 * 3072];
  __shared__ float sc[16 * 4 * 4];

  const int b   = blockIdx.x;
  const int tid = threadIdx.x;
  const size_t base = (size_t)b * 4 * 3072;

  for (int i = tid; i < (4 * 3072) / 8; i += 256)
    copy16_g2l(&qkvs[i * 8], &QKV[base + (size_t)i * 8]);
  async_wait_all();
  __syncthreads();

  // 256 threads = 16 heads * 4 q * 4 k score dots over dh=64
  {
    const int h = tid >> 4, qi = (tid >> 2) & 3, ki = tid & 3;
    const unsigned short* qp = &qkvs[qi * 3072 + h * 64];
    const unsigned short* kp = &qkvs[ki * 3072 + 1024 + h * 64];
    float s = 0.f;
#pragma unroll 8
    for (int d = 0; d < 64; ++d) s += bf2f(qp[d]) * bf2f(kp[d]);
    sc[(h * 4 + qi) * 4 + ki] = s * 0.125f;  // 1/sqrt(64)
  }
  __syncthreads();

  for (int i = tid; i < 4 * 1024; i += 256) {
    const int qi = i >> 10;
    const int dc = i & 1023;
    const int h  = dc >> 6;
    const float* sr = &sc[(h * 4 + qi) * 4];
    const float s0 = sr[0], s1 = sr[1], s2 = sr[2], s3 = sr[3];
    const float m  = fmaxf(fmaxf(s0, s1), fmaxf(s2, s3));
    const float e0 = __expf(s0 - m), e1 = __expf(s1 - m);
    const float e2 = __expf(s2 - m), e3 = __expf(s3 - m);
    const float inv = 1.f / (e0 + e1 + e2 + e3);
    const float c = e0 * bf2f(qkvs[0 * 3072 + 2048 + dc]) +
                    e1 * bf2f(qkvs[1 * 3072 + 2048 + dc]) +
                    e2 * bf2f(qkvs[2 * 3072 + 2048 + dc]) +
                    e3 * bf2f(qkvs[3 * 3072 + 2048 + dc]);
    CTX[(size_t)(b * 4 + qi) * 1024 + dc] = f2bf(c * inv);
  }
}

// ---------------------------------------------------------------------------
// x = rmsnorm(resid + addin); write fp32 (optionally in place) + optional bf16
// One block per 1024-wide row; 256 threads x 4 elems.
// ---------------------------------------------------------------------------
__global__ __launch_bounds__(256) void add_rmsnorm_kernel(
    const float* __restrict__ resid, const float* __restrict__ addin,
    float* __restrict__ xf32, unsigned short* __restrict__ xbf16) {
  __shared__ float red[256];
  const int row = blockIdx.x, tid = threadIdx.x;
  const size_t base = (size_t)row * 1024 + tid * 4;

  float4 a = *(const float4*)&resid[base];
  float4 c = *(const float4*)&addin[base];
  float t0 = a.x + c.x, t1 = a.y + c.y, t2 = a.z + c.z, t3 = a.w + c.w;
  red[tid] = t0 * t0 + t1 * t1 + t2 * t2 + t3 * t3;
  __syncthreads();
  for (int s = 128; s > 0; s >>= 1) {
    if (tid < s) red[tid] += red[tid + s];
    __syncthreads();
  }
  const float rms = rsqrtf(red[0] * (1.f / 1024.f) + 1e-6f);
  float4 o; o.x = t0 * rms; o.y = t1 * rms; o.z = t2 * rms; o.w = t3 * rms;
  *(float4*)&xf32[base] = o;
  if (xbf16) {
    xbf16[base + 0] = f2bf(o.x); xbf16[base + 1] = f2bf(o.y);
    xbf16[base + 2] = f2bf(o.z); xbf16[base + 3] = f2bf(o.w);
  }
}

// ---------------------------------------------------------------------------
extern "C" void kernel_launch(void* const* d_in, const int* in_sizes, int n_in,
                              void* d_out, int out_size, void* d_ws,
                              size_t ws_size, hipStream_t stream) {
  (void)in_sizes; (void)n_in; (void)out_size; (void)ws_size;
  constexpr int TOK = 16384 * 4;   // 65536 tokens
  constexpr int D   = 1024;
  constexpr int DH  = 4096;

  const float* seq   = (const float*)d_in[0];
  const float* w_in  = (const float*)d_in[1];
  const float* b_in  = (const float*)d_in[2];
  const float* w_out = (const float*)d_in[3];
  const float* b_out = (const float*)d_in[4];
  const float* w_g   = (const float*)d_in[5];
  const float* w_u   = (const float*)d_in[6];
  const float* w_d   = (const float*)d_in[7];

  // workspace layout (bytes)
  char* ws = (char*)d_ws;
  constexpr size_t OFF_WIN  = 0;                                   // 3072x1024 bf16
  constexpr size_t OFF_WOUT = OFF_WIN  + (size_t)3072 * 1024 * 2;  // 1024x1024 bf16
  constexpr size_t OFF_WG   = OFF_WOUT + (size_t)1024 * 1024 * 2;  // 4096x1024 bf16
  constexpr size_t OFF_WU   = OFF_WG   + (size_t)4096 * 1024 * 2;
  constexpr size_t OFF_WD   = OFF_WU   + (size_t)4096 * 1024 * 2;  // 1024x4096 bf16
  constexpr size_t OFF_R1   = OFF_WD   + (size_t)1024 * 4096 * 2;  // 128MB: xb -> ctx
  constexpr size_t OFF_R2   = OFF_R1   + (size_t)TOK * D * 2;      // 384MB: qkv -> {x2b, y}
  constexpr size_t OFF_R3   = OFF_R2   + (size_t)TOK * 3 * D * 2;  // 256MB: attn -> x_f32
  constexpr size_t OFF_R4   = OFF_R3   + (size_t)TOK * D * 4;      // 512MB: h bf16

  unsigned short* WB_in  = (unsigned short*)(ws + OFF_WIN);
  unsigned short* WB_out = (unsigned short*)(ws + OFF_WOUT);
  unsigned short* WB_g   = (unsigned short*)(ws + OFF_WG);
  unsigned short* WB_u   = (unsigned short*)(ws + OFF_WU);
  unsigned short* WB_d   = (unsigned short*)(ws + OFF_WD);
  unsigned short* XB     = (unsigned short*)(ws + OFF_R1);  // seq bf16
  unsigned short* CTX    = (unsigned short*)(ws + OFF_R1);  // reuse after GEMM1
  unsigned short* QKV    = (unsigned short*)(ws + OFF_R2);
  unsigned short* X2B    = (unsigned short*)(ws + OFF_R2);  // reuse after attention
  float*          Y      = (float*)(ws + OFF_R2 + (size_t)TOK * D * 2);
  float*          ATTN   = (float*)(ws + OFF_R3);
  float*          XF32   = (float*)(ws + OFF_R3);           // in-place norm
  unsigned short* H      = (unsigned short*)(ws + OFF_R4);

  const dim3 blk(256);

  // 1) fp32 -> bf16 conversions
  cvt_f32_to_bf16<<<2048, blk, 0, stream>>>(w_in,  WB_in,  3072 * 1024);
  cvt_f32_to_bf16<<<1024, blk, 0, stream>>>(w_out, WB_out, 1024 * 1024);
  cvt_f32_to_bf16<<<2048, blk, 0, stream>>>(w_g,   WB_g,   4096 * 1024);
  cvt_f32_to_bf16<<<2048, blk, 0, stream>>>(w_u,   WB_u,   4096 * 1024);
  cvt_f32_to_bf16<<<2048, blk, 0, stream>>>(w_d,   WB_d,   1024 * 4096);
  cvt_f32_to_bf16<<<4096, blk, 0, stream>>>(seq,   XB,     TOK * D);

  // 2) QKV projection: [65536,3072] bf16
  gemm_bf16_wmma<0><<<dim3(TOK / 64, 3 * D / 128), blk, 0, stream>>>(
      XB, D, WB_in, nullptr, D, QKV, 3 * D, b_in, 3 * D, D);

  // 3) attention -> ctx bf16
  attention_kernel<<<TOK / 4, blk, 0, stream>>>(QKV, CTX);

  // 4) out projection -> fp32
  gemm_bf16_wmma<1><<<dim3(TOK / 64, D / 128), blk, 0, stream>>>(
      CTX, D, WB_out, nullptr, D, ATTN, D, b_out, D, D);

  // 5) x = rmsnorm(seq + attn_out): fp32 in place + bf16 copy
  add_rmsnorm_kernel<<<TOK, blk, 0, stream>>>(seq, ATTN, XF32, X2B);

  // 6) fused SwiGLU gate/up: h = silu(x@Wg^T) * (x@Wu^T) -> bf16 [65536,4096]
  gemm_bf16_wmma<2><<<dim3(TOK / 64, DH / 128), blk, 0, stream>>>(
      X2B, D, WB_g, WB_u, D, H, DH, nullptr, DH, D);

  // 7) down projection -> fp32
  gemm_bf16_wmma<1><<<dim3(TOK / 64, D / 128), blk, 0, stream>>>(
      H, DH, WB_d, nullptr, DH, Y, D, nullptr, D, DH);

  // 8) out = rmsnorm(x + ffn)
  add_rmsnorm_kernel<<<TOK, blk, 0, stream>>>(XF32, Y, (float*)d_out, nullptr);
}
